// _LSTMBlock_40046275068570
// MI455X (gfx1250) — matrix operand order repaired and checked by reading, not compiled, not run on hardware
//
#include <hip/hip_runtime.h>

// ---------------------------------------------------------------------------
// Fused 2-layer LSTM (80 -> 100 -> 50), B=1024, T=512, batch_first, fp32 I/O.
// One block per 16 batch rows (one WMMA M-tile); weights LDS/register-resident
// as f16; per-step GEMMs via v_wmma_f32_16x16x32_f16 (f32 accumulate).
// Gate columns interleaved (col = 4*unit + gate): each N-tile owns complete
// hidden units -> activations are wave-local; 2 barriers per timestep.
// B-fragments double-buffered across tiles to hide LDS latency; activations
// use v_exp_f32 + v_rcp_f32 (no slow IEEE division).
// ---------------------------------------------------------------------------

typedef __attribute__((ext_vector_type(16))) _Float16 v16h;
typedef __attribute__((ext_vector_type(8)))  _Float16 v8h;
typedef __attribute__((ext_vector_type(8)))  float    v8f;
typedef __attribute__((ext_vector_type(4)))  float    v4f;

constexpr int BATCH = 1024;
constexpr int T     = 512;
constexpr int IN    = 80;
constexpr int H1    = 100;
constexpr int H2    = 50;

constexpr int K1  = 192;          // [x|h1|0]: 80+100 pad 192 (6 K-tiles)
constexpr int KT1 = K1 / 32;
constexpr int NG1 = 4 * H1;       // 400 interleaved gate cols = 25 N-tiles
constexpr int NT1 = NG1 / 16;
constexpr int K2  = 160;          // [h1|h2|0]: 100+50 pad 160 (5 K-tiles)
constexpr int KT2 = K2 / 32;
constexpr int NG2 = 208;          // 13 N-tiles (units 50,51 zero padding)
constexpr int NT2 = NG2 / 16;

constexpr int SSTR = 20;          // scratch row stride: LDS bank-conflict-free

__device__ __forceinline__ float sigf(float x) {
    // 1/(1+e^-x) with v_exp_f32 + v_rcp_f32
    float e = __expf(-x);
    return __builtin_amdgcn_rcpf(1.0f + e);
}
__device__ __forceinline__ float tanhfast(float x) {
    float e = __expf(-2.0f * x);
    return (1.0f - e) * __builtin_amdgcn_rcpf(1.0f + e);
}

// A fragment (16x32 f16): lanes 0-15 row M=lane hold K 0..15;
// lanes 16-31 hold K 16..31. Source row-major LDS.
__device__ __forceinline__ v16h load_a(const _Float16* base, int stride,
                                       int kt, int lane) {
    int m  = lane & 15;
    int kb = kt * 32 + ((lane >> 4) << 4);
    const _Float16* p = base + m * stride + kb;
    v8h lo = *(const v8h*)p;
    v8h hi = *(const v8h*)(p + 8);
    v16h a;
#pragma unroll
    for (int i = 0; i < 8; ++i) { a[i] = lo[i]; a[i + 8] = hi[i]; }
    return a;
}

// B fragment (32x16 f16): lane holds column N = nt*16 + lane%16, 16 contiguous
// K-halves at kt*32 + (lane/16)*16. Weights stored LDS-transposed w[n][k].
__device__ __forceinline__ v16h load_b(const _Float16* w, int stride,
                                       int nt, int kt, int lane) {
    int n  = nt * 16 + (lane & 15);
    int kb = kt * 32 + ((lane >> 4) << 4);
    return *(const v16h*)(w + n * stride + kb);
}

// ---- tile-body macros (macros, not functions: keep fragments in VGPRs) ----
#define LOADB1(BF, NT)                                                        \
    { _Pragma("unroll") for (int kt = 0; kt < KT1; ++kt)                      \
          BF[kt] = load_b(sW1, K1, (NT), kt, lane); }

#define L1_TILE(BF, CI, NT)                                                   \
    {                                                                         \
        v8f acc;                                                              \
        _Pragma("unroll") for (int v = 0; v < 8; ++v) acc[v] = b1r[CI];       \
        _Pragma("unroll") for (int kt = 0; kt < KT1; ++kt)                    \
            acc = __builtin_amdgcn_wmma_f32_16x16x32_f16(                     \
                false, a1[kt], false, BF[kt], (short)0, acc, false, false);   \
        _Pragma("unroll") for (int v = 0; v < 8; ++v)                         \
            ss[(mb + v) * SSTR + col] = acc[v];                               \
        _Pragma("unroll") for (int q = 0; q < 2; ++q) {                       \
            int u = (lane >> 4) + 2 * q;                                      \
            v4f g4 = *(const v4f*)(ss + r * SSTR + 4 * u);                    \
            float ig = sigf(g4[0]), fg = sigf(g4[1]);                         \
            float gg = tanhfast(g4[2]), og = sigf(g4[3]);                     \
            float c = fg * c1r[CI][q] + ig * gg;                              \
            c1r[CI][q] = c;                                                   \
            float h = og * tanhfast(c);                                       \
            int unit = 4 * (NT) + u;                                          \
            X1n[r * K1 + IN + unit] = (_Float16)h;                            \
            X2c[r * K2 + unit]      = (_Float16)h;                            \
        }                                                                     \
    }

#define L2_TILE(CI, NT)                                                       \
    {                                                                         \
        v8f acc;                                                              \
        _Pragma("unroll") for (int v = 0; v < 8; ++v) acc[v] = b2r[CI];       \
        _Pragma("unroll") for (int kt = 0; kt < KT2; ++kt)                    \
            acc = __builtin_amdgcn_wmma_f32_16x16x32_f16(                     \
                false, a2[kt], false, b2w[CI][kt], (short)0, acc,             \
                false, false);                                                \
        _Pragma("unroll") for (int v = 0; v < 8; ++v)                         \
            ss[(mb + v) * SSTR + col] = acc[v];                               \
        _Pragma("unroll") for (int q = 0; q < 2; ++q) {                       \
            int u = (lane >> 4) + 2 * q;                                      \
            v4f g4 = *(const v4f*)(ss + r * SSTR + 4 * u);                    \
            float ig = sigf(g4[0]), fg = sigf(g4[1]);                         \
            float gg = tanhfast(g4[2]), og = sigf(g4[3]);                     \
            float c = fg * c2r[CI][q] + ig * gg;                              \
            c2r[CI][q] = c;                                                   \
            float h = og * tanhfast(c);                                       \
            int unit = 4 * (NT) + u;                                          \
            if (unit < H2) {                                                  \
                X2n[r * K2 + H1 + unit] = (_Float16)h;                        \
                if (t == T - 1) out[(size_t)(b0 + r) * H2 + unit] = h;        \
            }                                                                 \
        }                                                                     \
    }

__global__ __launch_bounds__(256, 1)
void lstm2_fused(const float* __restrict__ x,
                 const float* __restrict__ Wih1, const float* __restrict__ Whh1,
                 const float* __restrict__ bih1, const float* __restrict__ bhh1,
                 const float* __restrict__ Wih2, const float* __restrict__ Whh2,
                 const float* __restrict__ bih2, const float* __restrict__ bhh2,
                 float* __restrict__ out) {
    // ---- LDS (~250 KB of 320 KB) ----
    __shared__ __align__(32) _Float16 sW1[NG1 * K1];     // 150 KB
    __shared__ __align__(32) _Float16 sW2[NG2 * K2];     //  65 KB
    __shared__ __align__(32) _Float16 sX1[2 * 16 * K1];  //  12 KB  ping-pong [x|h1|0]
    __shared__ __align__(32) _Float16 sX2[2 * 16 * K2];  //  10 KB  ping-pong [h1|h2|0]
    __shared__ __align__(16) float    sS[8 * 16 * SSTR]; //  10 KB  per-wave scratch
    __shared__ float sB1[NG1];
    __shared__ float sB2[NG2];

    const int tid  = threadIdx.x;
    const int lane = tid & 31;
    const int wave = __builtin_amdgcn_readfirstlane(tid >> 5);  // scalar
    const int b0   = blockIdx.x * 16;

    // ---- one-time staging: f16 weights, transposed + gate-interleaved ----
    for (int i = tid; i < NG1 * K1; i += 256) {
        int n = i / K1, k = i - n * K1;
        int unit = n >> 2, gate = n & 3;
        int orow = gate * H1 + unit;
        float w = (k < IN)      ? Wih1[orow * IN + k]
                : (k < IN + H1) ? Whh1[orow * H1 + (k - IN)]
                                : 0.0f;
        sW1[i] = (_Float16)w;
    }
    for (int i = tid; i < NG2 * K2; i += 256) {
        int n = i / K2, k = i - n * K2;
        int unit = n >> 2, gate = n & 3;
        float w = 0.0f;
        if (unit < H2) {
            int orow = gate * H2 + unit;
            if (k < H1)           w = Wih2[orow * H1 + k];
            else if (k < H1 + H2) w = Whh2[orow * H2 + (k - H1)];
        }
        sW2[i] = (_Float16)w;
    }
    for (int n = tid; n < NG1; n += 256) {
        int unit = n >> 2, gate = n & 3, orow = gate * H1 + unit;
        sB1[n] = bih1[orow] + bhh1[orow];
    }
    for (int n = tid; n < NG2; n += 256) {
        int unit = n >> 2, gate = n & 3;
        sB2[n] = (unit < H2) ? (bih2[gate * H2 + unit] + bhh2[gate * H2 + unit])
                             : 0.0f;
    }
    for (int i = tid; i < 2 * 16 * K1; i += 256) sX1[i] = (_Float16)0.0f;
    for (int i = tid; i < 2 * 16 * K2; i += 256) sX2[i] = (_Float16)0.0f;
    __syncthreads();

    // ---- per-wave register state (constant wave->tile ownership) ----
    // Tiles: L1 = {wave, wave+8, wave+16} all waves, + {24} for wave 0.
    //        L2 = {wave} all waves, + {wave+8} for wave < 5.
    float c1r[4][2], c2r[2][2];
    float b1r[4], b2r[2];
    v16h  b2w[2][KT2];               // layer-2 B fragments live in VGPRs
#pragma unroll
    for (int i = 0; i < 4; ++i) {
        c1r[i][0] = c1r[i][1] = 0.0f;
        int nt = wave + 8 * i;
        b1r[i] = (nt < NT1) ? sB1[nt * 16 + (lane & 15)] : 0.0f;
    }
#pragma unroll
    for (int i = 0; i < 2; ++i) {
        c2r[i][0] = c2r[i][1] = 0.0f;
        int nt = wave + 8 * i;
        b2r[i] = (nt < NT2) ? sB2[nt * 16 + (lane & 15)] : 0.0f;
#pragma unroll
        for (int kt = 0; kt < KT2; ++kt)
            b2w[i][kt] = (nt < NT2) ? load_b(sW2, K2, nt, kt, lane)
                                    : (v16h)(_Float16)0.0f;
    }
    float* ss = sS + wave * 16 * SSTR;
    const int col = lane & 15;
    const int mb  = (lane >> 4) << 3;
    const int r   = lane & 15;

    // ---- sequential scan ----
    for (int t = 0; t < T; ++t) {
        _Float16* X1c = sX1 + (t & 1) * 16 * K1;
        _Float16* X1n = sX1 + ((t + 1) & 1) * 16 * K1;
        _Float16* X2c = sX2 + (t & 1) * 16 * K2;
        _Float16* X2n = sX2 + ((t + 1) & 1) * 16 * K2;

        // (A) stage x_t (f32->f16) into X1c[:,0:80]; 16 threads per row
        {
            int xr = tid >> 4, cb = tid & 15;
            const float* xrow = x + ((size_t)(b0 + xr) * T + t) * IN;
#pragma unroll
            for (int j = 0; j < 5; ++j)
                X1c[xr * K1 + cb + 16 * j] = (_Float16)xrow[cb + 16 * j];
        }
        if (t + 1 < T && tid < 64) {   // prefetch next x tile into caches
            int m = tid >> 2, part = tid & 3;
            __builtin_prefetch(
                x + ((size_t)(b0 + m) * T + (t + 1)) * IN + part * 20, 0, 3);
        }
        __syncthreads();   // barrier 1: x_t staged, prev h1/h2 visible

        // (B) layer-1 GEMM + wave-local activations, B double-buffered
        v16h a1[KT1];
#pragma unroll
        for (int kt = 0; kt < KT1; ++kt) a1[kt] = load_a(X1c, K1, kt, lane);
        {
            v16h bA[KT1], bB[KT1];
            LOADB1(bA, wave);            // tile 0 fragments
            LOADB1(bB, wave + 8);        // tile 1 fragments in flight
            L1_TILE(bA, 0, wave);
            LOADB1(bA, wave + 16);       // tile 2 fragments in flight
            L1_TILE(bB, 1, wave + 8);
            if (wave == 0) LOADB1(bB, 24);   // scalar branch, EXEC untouched
            L1_TILE(bA, 2, wave + 16);
            if (wave == 0) L1_TILE(bB, 3, 24);
        }
        __syncthreads();   // barrier 2: h1(t) complete in X2c

        // (C) layer-2 GEMM (B cached in registers) + wave-local activations
        v16h a2[KT2];
#pragma unroll
        for (int kt = 0; kt < KT2; ++kt) a2[kt] = load_a(X2c, K2, kt, lane);
        L2_TILE(0, wave);
        if (wave < 5) L2_TILE(1, wave + 8);
        // no barrier: barrier 1 of t+1 orders h2 writes before readers
    }
}

extern "C" void kernel_launch(void* const* d_in, const int* in_sizes, int n_in,
                              void* d_out, int out_size, void* d_ws, size_t ws_size,
                              hipStream_t stream) {
    const float* x    = (const float*)d_in[0];
    const float* Wih1 = (const float*)d_in[1];
    const float* Whh1 = (const float*)d_in[2];
    const float* bih1 = (const float*)d_in[3];
    const float* bhh1 = (const float*)d_in[4];
    const float* Wih2 = (const float*)d_in[5];
    const float* Whh2 = (const float*)d_in[6];
    const float* bih2 = (const float*)d_in[7];
    const float* bhh2 = (const float*)d_in[8];
    float* out = (float*)d_out;

    dim3 grid(BATCH / 16);   // 64 blocks, one 16-row M-tile each
    dim3 block(256);         // 8 wave32s
    lstm2_fused<<<grid, block, 0, stream>>>(x, Wih1, Whh1, bih1, bhh1,
                                            Wih2, Whh2, bih2, bhh2, out);
}
